// RefineDetMultiBoxLoss_80376017977632
// MI455X (gfx1250) — compile-verified, matrix operand order checked by reading, not run on hardware
//
#include <hip/hip_runtime.h>
#include <hip/hip_bf16.h>
#include <math.h>

#define THRV   0.5f
#define VAR0   0.1f
#define VAR1   0.2f
#define RATIO  3
#define MAXN   16
#define MAXC   21
#define TPB_A  256
#define TPB_B  128
#define TPB_C  256
#define CHUNK  (TPB_C * 32)     // predicates counted per block pass

typedef int v8i __attribute__((ext_vector_type(8)));
typedef int v4i_g __attribute__((vector_size(16)));   // matches builtin param pointee
typedef __attribute__((address_space(1))) v4i_g* g_v4i_p;
typedef __attribute__((address_space(3))) v4i_g* l_v4i_p;
typedef __attribute__((address_space(1))) int*   g_i_p;
typedef __attribute__((address_space(3))) int*   l_i_p;

#if defined(__has_builtin)
#  if __has_builtin(__builtin_amdgcn_global_load_async_to_lds_b128) && \
      __has_builtin(__builtin_amdgcn_global_load_async_to_lds_b32) &&  \
      __has_builtin(__builtin_amdgcn_s_wait_asynccnt)
#    define HAVE_ASYNC_LDS 1
#  endif
#endif
#ifndef HAVE_ASYNC_LDS
#  define HAVE_ASYNC_LDS 0
#endif

// ---------------------------------------------------------------------------
// Kernel 0: zero per-batch accumulators (harness does not re-zero between runs)
// ---------------------------------------------------------------------------
__global__ void init_kernel(int* num_pos, float* loss_l, float* ce_pos,
                            float* neg_sum, int B) {
    int i = blockIdx.x * blockDim.x + threadIdx.x;
    if (i < B) { num_pos[i] = 0; loss_l[i] = 0.f; ce_pos[i] = 0.f; neg_sum[i] = 0.f; }
}

// ---------------------------------------------------------------------------
// Kernel 1: per-batch matching.
//   bto[b,p] = max_n IoU, bti[b,p] = argmax_n (first max),
//   bpi[b,n] = argmax_p IoU (first max, i.e. smallest p among ties).
// ---------------------------------------------------------------------------
__global__ void __launch_bounds__(TPB_A)
match_kernel(const float* __restrict__ priors, const float* __restrict__ truths,
             int P, int N, float* __restrict__ bto, int* __restrict__ bti,
             int* __restrict__ bpi) {
    __shared__ float s_tr[MAXN * 4];
    __shared__ float s_val[TPB_A];
    __shared__ int   s_idx[TPB_A];
    const int b = blockIdx.x;
    const int t = threadIdx.x;
    if (t < N * 4) s_tr[t] = truths[(size_t)b * N * 4 + t];
    __syncthreads();

    float bestv[MAXN]; int besti[MAXN];
    for (int n = 0; n < N; ++n) { bestv[n] = -1.f; besti[n] = 0; }

    for (int p = t; p < P; p += TPB_A) {
        const float px = priors[p*4+0], py = priors[p*4+1];
        const float pw = priors[p*4+2], ph = priors[p*4+3];
        const float bx1 = px - 0.5f*pw, by1 = py - 0.5f*ph;
        const float bx2 = px + 0.5f*pw, by2 = py + 0.5f*ph;
        const float areab = pw * ph;
        float pbv = -1.f; int pbn = 0;
        for (int n = 0; n < N; ++n) {
            const float ax1 = s_tr[n*4+0], ay1 = s_tr[n*4+1];
            const float ax2 = s_tr[n*4+2], ay2 = s_tr[n*4+3];
            const float iw = fmaxf(fminf(ax2, bx2) - fmaxf(ax1, bx1), 0.f);
            const float ih = fmaxf(fminf(ay2, by2) - fmaxf(ay1, by1), 0.f);
            const float inter = iw * ih;
            const float areaa = (ax2 - ax1) * (ay2 - ay1);
            const float ov = inter / (areaa + areab - inter);
            if (ov > bestv[n]) { bestv[n] = ov; besti[n] = p; }   // p ascending -> first max
            if (ov > pbv)      { pbv = ov;      pbn = n; }        // n ascending -> first max
        }
        bto[(size_t)b*P + p] = pbv;
        bti[(size_t)b*P + p] = pbn;
    }

    // block argmax per truth, tie -> smaller prior index
    for (int n = 0; n < N; ++n) {
        s_val[t] = bestv[n]; s_idx[t] = besti[n];
        __syncthreads();
        for (int s = TPB_A / 2; s > 0; s >>= 1) {
            if (t < s) {
                const float v2 = s_val[t + s]; const int i2 = s_idx[t + s];
                if (v2 > s_val[t] || (v2 == s_val[t] && i2 < s_idx[t])) {
                    s_val[t] = v2; s_idx[t] = i2;
                }
            }
            __syncthreads();
        }
        if (t == 0) bpi[b * N + n] = s_idx[0];
        __syncthreads();
    }
}

// ---------------------------------------------------------------------------
// Kernel 2: per-prior targets, fused lse/ce/smooth-L1.
//   conf tile (128 priors x 21 classes) staged into LDS via async loads.
// ---------------------------------------------------------------------------
__global__ void __launch_bounds__(TPB_B)
targets_kernel(const float* __restrict__ loc, const float* __restrict__ conf,
               const float* __restrict__ priors, const float* __restrict__ truths,
               const int* __restrict__ labels, const float* __restrict__ bto,
               const int* __restrict__ bti, const int* __restrict__ bpi,
               int P, int Cc, int N,
               float* __restrict__ lc, int* num_pos, float* loss_l_acc,
               float* ce_pos_acc) {
    __shared__ float s_conf[TPB_B * MAXC];
    __shared__ float s_tr[MAXN * 4];
    __shared__ int   s_lab[MAXN];
    __shared__ int   s_bpi[MAXN];
    __shared__ float s_red[TPB_B];
    __shared__ int   s_redi[TPB_B];

    const int b  = blockIdx.y;
    const int p0 = blockIdx.x * TPB_B;
    const int t  = threadIdx.x;
    const int cnt = min(TPB_B, P - p0);

    if (t < N * 4) s_tr[t] = truths[(size_t)b * N * 4 + t];
    if (t < N) { s_lab[t] = labels[b * N + t]; s_bpi[t] = bpi[b * N + t]; }

    const float* gsrc = conf + ((size_t)b * P + p0) * Cc;
    const int nd = cnt * Cc;                       // dwords in this tile
#if HAVE_ASYNC_LDS
    {
        const int n128 = nd >> 2;                  // 16B chunks (tile is 16B aligned)
        for (int i = t; i < n128; i += TPB_B) {
            __builtin_amdgcn_global_load_async_to_lds_b128(
                (g_v4i_p)(gsrc + 4 * i),
                (l_v4i_p)(&s_conf[4 * i]), 0, 0);
        }
        for (int i = (n128 << 2) + t; i < nd; i += TPB_B) {
            __builtin_amdgcn_global_load_async_to_lds_b32(
                (g_i_p)(gsrc + i),
                (l_i_p)(&s_conf[i]), 0, 0);
        }
        __builtin_amdgcn_s_wait_asynccnt(0);
    }
#else
    for (int i = t; i < nd; i += TPB_B) s_conf[i] = gsrc[i];
#endif
    __syncthreads();

    float my_lsum = 0.f, my_ce = 0.f; int my_pos = 0;
    if (t < cnt) {
        const int p = p0 + t;
        const size_t bp = (size_t)b * P + p;
        const float* row = &s_conf[t * Cc];
        float m = row[0];
        for (int c = 1; c < Cc; ++c) m = fmaxf(m, row[c]);
        float se = 0.f;
        for (int c = 0; c < Cc; ++c) se += expf(row[c] - m);
        const float lse = m + logf(se);

        int idx = bti[bp]; float ovl = bto[bp];
        for (int n = 0; n < N; ++n) if (s_bpi[n] == p) { idx = n; ovl = 2.f; }
        const int ct = (ovl < THRV) ? 0 : (s_lab[idx] + 1);

        if (ct > 0) {
            const float ax1 = s_tr[idx*4+0], ay1 = s_tr[idx*4+1];
            const float ax2 = s_tr[idx*4+2], ay2 = s_tr[idx*4+3];
            const float px = priors[p*4+0], py = priors[p*4+1];
            const float pw = priors[p*4+2], ph = priors[p*4+3];
            float g[4];
            g[0] = (0.5f * (ax1 + ax2) - px) / (VAR0 * pw);
            g[1] = (0.5f * (ay1 + ay2) - py) / (VAR0 * ph);
            g[2] = logf((ax2 - ax1) / pw) / VAR1;
            g[3] = logf((ay2 - ay1) / ph) / VAR1;
            const float* ld = loc + bp * 4;
            #pragma unroll
            for (int j = 0; j < 4; ++j) {
                const float d = ld[j] - g[j], ad = fabsf(d);
                my_lsum += (ad < 1.f) ? 0.5f * d * d : (ad - 0.5f);
            }
            my_pos = 1;
            my_ce = lse - row[ct];
            lc[bp] = 0.f;                    // positives zeroed in mining proxy
        } else {
            lc[bp] = lse - row[0];           // ce for background == mining proxy
        }
    }

    s_red[t] = my_lsum; s_redi[t] = my_pos;
    __syncthreads();
    for (int s = TPB_B / 2; s; s >>= 1) {
        if (t < s) { s_red[t] += s_red[t + s]; s_redi[t] += s_redi[t + s]; }
        __syncthreads();
    }
    if (t == 0) {
        if (s_red[0] != 0.f) atomicAdd(&loss_l_acc[b], s_red[0]);
        if (s_redi[0])       atomicAdd(&num_pos[b], s_redi[0]);
    }
    __syncthreads();
    s_red[t] = my_ce;
    __syncthreads();
    for (int s = TPB_B / 2; s; s >>= 1) {
        if (t < s) s_red[t] += s_red[t + s];
        __syncthreads();
    }
    if (t == 0 && s_red[0] != 0.f) atomicAdd(&ce_pos_acc[b], s_red[0]);
}

// ---------------------------------------------------------------------------
// Kernel 3: per-batch top-k hard-negative sum.
//   count(lc > thr) via v_wmma_i32_16x16x64_iu8: B := all-ones, so
//   D[m,n] = sum_k A[m,k]; total = lane0 + lane16 accumulator sums.
//   Loads are UNCONDITIONAL aligned float4 (the lc buffer is padded by one
//   CHUNK), bounds apply only to the predicate byte -> no guarded scalar
//   loads / per-element waits inside the hot loop.
// ---------------------------------------------------------------------------
__device__ __forceinline__ int block_count_gt(const float* __restrict__ L, int P,
                                              unsigned thr, int t, int* s_cnt) {
    union PK { v8i v; unsigned char b[32]; };
    v8i acc = {0, 0, 0, 0, 0, 0, 0, 0};
    const v8i ones = {0x01010101, 0x01010101, 0x01010101, 0x01010101,
                      0x01010101, 0x01010101, 0x01010101, 0x01010101};
    for (int base = 0; base < P; base += CHUNK) {
        const int start = base + t * 32;          // 128B-aligned
        __builtin_prefetch(&L[start + CHUNK], 0, 3);
        PK pk;
        #pragma unroll
        for (int q = 0; q < 8; ++q) {
            const float4 v = *(const float4*)&L[start + 4 * q];  // always in-bounds (padded)
            const int pbase = start + 4 * q;
            pk.b[4*q+0] = (unsigned char)((pbase + 0 < P) & (__float_as_uint(v.x) > thr));
            pk.b[4*q+1] = (unsigned char)((pbase + 1 < P) & (__float_as_uint(v.y) > thr));
            pk.b[4*q+2] = (unsigned char)((pbase + 2 < P) & (__float_as_uint(v.z) > thr));
            pk.b[4*q+3] = (unsigned char)((pbase + 3 < P) & (__float_as_uint(v.w) > thr));
        }
        acc = __builtin_amdgcn_wmma_i32_16x16x64_iu8(false, pk.v, false, ones,
                                                     acc, false, false);
    }
    int s = 0;
    #pragma unroll
    for (int v = 0; v < 8; ++v) s += acc[v];
    const int wave = t >> 5, lane = t & 31;
    const int wtot = __shfl(s, 0, 32) + __shfl(s, 16, 32);  // rows 0-7 + rows 8-15
    if (lane == 0) s_cnt[wave] = wtot;
    __syncthreads();
    int tot = 0;
    #pragma unroll
    for (int w = 0; w < TPB_C / 32; ++w) tot += s_cnt[w];
    __syncthreads();
    return tot;
}

__global__ void __launch_bounds__(TPB_C)
topk_kernel(const float* __restrict__ lc, const int* __restrict__ num_pos,
            int P, float* __restrict__ neg_sum) {
    __shared__ int   s_cnt[TPB_C / 32];
    __shared__ float s_red[TPB_C];
    const int b = blockIdx.x, t = threadIdx.x;
    const float* L = lc + (size_t)b * P;
    const int k = min(RATIO * num_pos[b], P - 1);
    if (k <= 0) { if (t == 0) neg_sum[b] = 0.f; return; }

    // lower_bound: smallest u with count(bits > u) < k   (lc >= 0 -> uint order)
    unsigned lo = 0u, hi = 0x7f800000u;
    while (lo < hi) {
        const unsigned mid = lo + ((hi - lo) >> 1);
        const int c = block_count_gt(L, P, mid, t, s_cnt);
        if (c < k) hi = mid; else lo = mid + 1;
    }
    const unsigned u = lo;                         // bits of k-th largest value
    const int cgt = block_count_gt(L, P, u, t, s_cnt);

    float fs = 0.f;
    for (int p = t; p < P; p += TPB_C) {
        const float v = L[p];
        if (__float_as_uint(v) > u) fs += v;
    }
    s_red[t] = fs;
    __syncthreads();
    for (int s = TPB_C / 2; s; s >>= 1) {
        if (t < s) s_red[t] += s_red[t + s];
        __syncthreads();
    }
    if (t == 0)
        neg_sum[b] = s_red[0] + (float)(k - cgt) * __uint_as_float(u);
}

// ---------------------------------------------------------------------------
// Kernel 4: finalize two scalars
// ---------------------------------------------------------------------------
__global__ void finalize_kernel(const int* num_pos, const float* loss_l,
                                const float* ce_pos, const float* neg_sum,
                                int B, float* out) {
    if (threadIdx.x == 0 && blockIdx.x == 0) {
        int tp = 0; float ll = 0.f, cc = 0.f;
        for (int b = 0; b < B; ++b) {
            tp += num_pos[b];
            ll += loss_l[b];
            cc += ce_pos[b] + neg_sum[b];
        }
        const float tot = (float)tp;
        out[0] = ll / tot;
        out[1] = cc / tot;
    }
}

// ---------------------------------------------------------------------------
extern "C" void kernel_launch(void* const* d_in, const int* in_sizes, int n_in,
                              void* d_out, int out_size, void* d_ws, size_t ws_size,
                              hipStream_t stream) {
    const float* loc    = (const float*)d_in[0];
    const float* conf   = (const float*)d_in[1];
    const float* priors = (const float*)d_in[2];
    const float* truths = (const float*)d_in[3];
    const int*   labels = (const int*)d_in[4];

    const int P  = in_sizes[2] / 4;
    const int B  = in_sizes[0] / (4 * P);
    const int Cc = in_sizes[1] / (B * P);
    const int N  = in_sizes[4] / B;

    char* ws = (char*)d_ws;
    size_t off = 0;
    auto carve = [&](size_t bytes) -> char* {
        char* p = ws + off;
        off = (off + bytes + 255) & ~(size_t)255;
        return p;
    };
    float* bto       = (float*)carve((size_t)B * P * sizeof(float));
    int*   bti       = (int*)  carve((size_t)B * P * sizeof(int));
    // padded by one CHUNK so the counting loop may load unconditionally
    float* lcv       = (float*)carve(((size_t)B * P + CHUNK) * sizeof(float));
    int*   bpi       = (int*)  carve((size_t)B * N * sizeof(int));
    int*   d_npos    = (int*)  carve((size_t)B * sizeof(int));
    float* d_lossl   = (float*)carve((size_t)B * sizeof(float));
    float* d_cepos   = (float*)carve((size_t)B * sizeof(float));
    float* d_negsum  = (float*)carve((size_t)B * sizeof(float));

    init_kernel<<<(B + 63) / 64, 64, 0, stream>>>(d_npos, d_lossl, d_cepos,
                                                  d_negsum, B);
    match_kernel<<<B, TPB_A, 0, stream>>>(priors, truths, P, N, bto, bti, bpi);

    dim3 gB((P + TPB_B - 1) / TPB_B, B);
    targets_kernel<<<gB, TPB_B, 0, stream>>>(loc, conf, priors, truths, labels,
                                             bto, bti, bpi, P, Cc, N, lcv,
                                             d_npos, d_lossl, d_cepos);

    topk_kernel<<<B, TPB_C, 0, stream>>>(lcv, d_npos, P, d_negsum);

    finalize_kernel<<<1, 32, 0, stream>>>(d_npos, d_lossl, d_cepos, d_negsum, B,
                                          (float*)d_out);
}
